// SqrllLayer_58591943852813
// MI455X (gfx1250) — matrix operand order repaired and checked by reading, not compiled, not run on hardware
//
#include <hip/hip_runtime.h>
#include <hip/hip_bf16.h>
#include <stddef.h>

// ---------------------------------------------------------------------------
// SQRLL layer for MI455X (gfx1250, wave32, WMMA, async-LDS).
//  * GEMMs use v_wmma_f32_16x16x32_bf16 (f32 accumulate).
//  * Gate GEMM: bf16 pre-converted inputs, quad-buffered LDS tiles fed by
//    global_load_async_to_lds_b128 (ASYNCcnt), one barrier per K-stage,
//    2 M-tiles per wave so each LDS B-fragment feeds two WMMAs.
//    All buffer indices are compile-time literals (power-of-2 buffer count +
//    fully unrolled 4-stage groups) so no dynamic VGPR indexing is emitted.
//  * T=2048 recurrence -> 3-pass chunked linear scan (131K-way parallel).
// ---------------------------------------------------------------------------

typedef __bf16 bf16_t;
typedef __attribute__((ext_vector_type(16))) __bf16 v16bf;
typedef __attribute__((ext_vector_type(8)))  __bf16 bf16x8;
typedef __attribute__((ext_vector_type(4)))  __bf16 bf16x4;
typedef __attribute__((ext_vector_type(8)))  float   v8f;

#define B_    8
#define T_    2048
#define NIN_  1024
#define NM_   1024
#define NOUT_ 1024
#define MT_   (B_ * T_)          // 16384 GEMM rows

#define NCHUNK 16
#define CHUNK  (T_ / NCHUNK)     // 128

__device__ __forceinline__ float sigmoidf_(float v) {
  return 1.0f / (1.0f + __expf(-v));
}

// Truncating a generic pointer to an LDS object yields the LDS byte offset
// (ISA 10.2: offset = low 32 bits of the address).
__device__ __forceinline__ unsigned lds_off(const void* p) {
  return (unsigned)(unsigned long long)p;
}

// One wave-wide async copy: each lane moves 16 B global->LDS. ASYNCcnt++.
__device__ __forceinline__ void async_ld16(unsigned ldsaddr, unsigned long long gaddr) {
  asm volatile("global_load_async_to_lds_b128 %0, %1, off"
               :: "v"(ldsaddr), "v"(gaddr) : "memory");
}

// --- WMMA fragment helpers (layouts per CDNA5 ISA 7.12.2, wave32) -----------
__device__ __forceinline__ v16bf load_frag_a(const bf16_t* row, int hlf) {
  bf16x8 lo = *(const bf16x8*)(row + hlf * 8);        // K = 8h .. 8h+7
  bf16x8 hi = *(const bf16x8*)(row + 16 + hlf * 8);   // K = 16+8h .. +7
  return __builtin_shufflevector(lo, hi, 0,1,2,3,4,5,6,7,8,9,10,11,12,13,14,15);
}
__device__ __forceinline__ v16bf load_frag_b(const bf16_t* row, int hlf) {
  bf16x8 lo = *(const bf16x8*)(row + hlf * 8);
  bf16x8 hi = *(const bf16x8*)(row + 16 + hlf * 8);
  return __builtin_shufflevector(lo, hi, 0,1,2,3,4,5,6,7,8,9,10,11,12,13,14,15);
}

// ===========================================================================
// Elementwise f32 -> bf16 (vectorized float4 -> bf16x4)
// ===========================================================================
__global__ void cvt_f32_to_bf16(const float* __restrict__ src,
                                bf16_t* __restrict__ dst, int n4)
{
  const int i = blockIdx.x * blockDim.x + threadIdx.x;
  if (i < n4) {
    const float4 v = ((const float4*)src)[i];
    bf16x4 o = { (bf16_t)v.x, (bf16_t)v.y, (bf16_t)v.z, (bf16_t)v.w };
    ((bf16x4*)dst)[i] = o;
  }
}

// ===========================================================================
// Kernel 1: fused gate GEMMs on bf16 inputs, async-LDS quad buffered.
// Block tile 256M x 32N, K step 32.  Wave tile: 32M x 32N x 4 gates
// (16 WMMA / 20 LDS-b128 per stage).  Per wave per stage: 6 async b128 ops
// (4 X row-groups + 2 weight row-groups), so steady-state wait is <=6.
// ===========================================================================
#define GA_BM 256
#define GA_BN 32
#define GA_BK 32
#define GA_PITCH 40      // bf16 elems per LDS row: 32 data + 8 pad (80 B)
#define GA_NK   (NIN_ / GA_BK)      // 32 K-stages
#define GA_NBUF 4
#define GA_XLD  4        // async instrs per wave per stage for X
#define GA_WLD  2        // async instrs per wave per stage for W
#define GA_XBUFB (GA_BM * GA_PITCH * 2)       // 20480 B per X buffer
#define GA_WBUFB (4 * GA_BN * GA_PITCH * 2)   // 10240 B per W buffer

__launch_bounds__(256)
__global__ void sqrll_gates(const bf16_t* __restrict__ xb,
                            const bf16_t* __restrict__ wb,   // [4][NM_][NIN_] bf16
                            const float* __restrict__ wr_b,
                            const float* __restrict__ wig_b,
                            const float* __restrict__ wog_b,
                            float* __restrict__ xi_o, float* __restrict__ r_o,
                            float* __restrict__ og_o)
{
  __shared__ __align__(16) bf16_t Xs[GA_NBUF][GA_BM * GA_PITCH];
  __shared__ __align__(16) bf16_t Ws[GA_NBUF][4][GA_BN * GA_PITCH];

  const int m0   = blockIdx.x * GA_BM;
  const int n0   = blockIdx.y * GA_BN;
  const int tid  = threadIdx.x;
  const int wv   = tid >> 5;          // 8 waves, wave -> 32 M rows
  const int lane = tid & 31;
  const int hlf  = lane >> 4;
  const int l16  = lane & 15;
  const int l4   = lane & 3;          // 16-B column within a 64-B row chunk
  const int l8   = lane >> 2;         // sub-row within an 8-row async group

  // ---- per-thread async source/dest addressing (buffer-0 LDS offsets) ------
  unsigned long long gX[GA_XLD], gW[GA_WLD];
  unsigned ldsX0[GA_XLD], ldsW0[GA_WLD];
  {
    const unsigned xs0 = lds_off(&Xs[0][0]);
    const unsigned ws0 = lds_off(&Ws[0][0][0]);
    #pragma unroll
    for (int i = 0; i < GA_XLD; ++i) {
      const int xrow = wv * 32 + i * 8 + l8;                   // 0..255
      gX[i] = (unsigned long long)(xb + (size_t)(m0 + xrow) * NIN_) + (unsigned)(l4 * 16);
      ldsX0[i] = xs0 + (unsigned)(xrow * GA_PITCH) * 2 + l4 * 16;
    }
    #pragma unroll
    for (int j = 0; j < GA_WLD; ++j) {
      const int idx  = wv * 2 + j;                             // 0..15
      const int g    = idx >> 2;                               // gate
      const int wrow = (idx & 3) * 8 + l8;                     // 0..31
      gW[j] = (unsigned long long)(wb + ((size_t)g * NM_ + n0 + wrow) * NIN_) + (unsigned)(l4 * 16);
      ldsW0[j] = ws0 + (unsigned)(g * (GA_BN * GA_PITCH) + wrow * GA_PITCH) * 2 + l4 * 16;
    }
  }

  v8f acc[4][2][2] = {};              // [gate][n-tile][m-tile]

  // issue one K-stage into buffer `buf` (buf is always a literal constant)
  auto issue_stage = [&](int buf) {
    const unsigned xoff = (unsigned)(buf * GA_XBUFB);
    const unsigned woff = (unsigned)(buf * GA_WBUFB);
    #pragma unroll
    for (int i = 0; i < GA_XLD; ++i) { async_ld16(ldsX0[i] + xoff, gX[i]); gX[i] += 64ull; }
    #pragma unroll
    for (int j = 0; j < GA_WLD; ++j) { async_ld16(ldsW0[j] + woff, gW[j]); gW[j] += 64ull; }
  };

  // compute one K-stage from buffer `cur` (cur is always a literal constant)
  auto compute_stage = [&](int cur) {
    const v16bf a0 = load_frag_a(&Xs[cur][(wv * 32 +      l16) * GA_PITCH], hlf);
    const v16bf a1 = load_frag_a(&Xs[cur][(wv * 32 + 16 + l16) * GA_PITCH], hlf);
    #pragma unroll
    for (int g = 0; g < 4; ++g) {
      #pragma unroll
      for (int j = 0; j < 2; ++j) {
        const v16bf b = load_frag_b(&Ws[cur][g][(j * 16 + l16) * GA_PITCH], hlf);
        acc[g][j][0] = __builtin_amdgcn_wmma_f32_16x16x32_bf16(
            false, a0, false, b, (short)0, acc[g][j][0], false, false);
        acc[g][j][1] = __builtin_amdgcn_wmma_f32_16x16x32_bf16(
            false, a1, false, b, (short)0, acc[g][j][1], false, false);
      }
    }
  };

  issue_stage(0);                     // prefetch stages 0 and 1
  issue_stage(1);

  // Steady state: stages 0..27 in groups of 4; all indices literal.
  // Wait <=6: oldest stage landed (in-order completion), next stays in flight.
  // Buffer (kb+2)&3 held stage kb-2, finished by all waves before barrier kb.
  #pragma unroll 1
  for (int kt = 0; kt < GA_NK / 4 - 1; ++kt) {
    #pragma unroll
    for (int s = 0; s < 4; ++s) {
      asm volatile("s_wait_asynccnt 0x6" ::: "memory");
      __syncthreads();
      compute_stage(s);
      issue_stage((s + 2) & 3);
    }
  }
  // Peeled tail: stages 28..31 (issues for 30, 31 only; then drain).
  asm volatile("s_wait_asynccnt 0x6" ::: "memory"); __syncthreads();
  compute_stage(0); issue_stage(2);
  asm volatile("s_wait_asynccnt 0x6" ::: "memory"); __syncthreads();
  compute_stage(1); issue_stage(3);
  asm volatile("s_wait_asynccnt 0x6" ::: "memory"); __syncthreads();
  compute_stage(2);
  asm volatile("s_wait_asynccnt 0x0" ::: "memory"); __syncthreads();
  compute_stage(3);

  // Epilogue: element v of acc[g][j][mt]: M = wv*32 + mt*16 + 8*hlf + v,
  //           N = j*16 + l16
  #pragma unroll
  for (int j = 0; j < 2; ++j) {
    const int n = n0 + j * 16 + l16;
    const float rb = wr_b[n], igb = wig_b[n], ogb = wog_b[n];
    #pragma unroll
    for (int mt = 0; mt < 2; ++mt) {
      #pragma unroll
      for (int v = 0; v < 8; ++v) {
        const int m = m0 + wv * 32 + mt * 16 + hlf * 8 + v;
        const size_t off = (size_t)m * NM_ + n;
        r_o[off]  = sigmoidf_(acc[0][j][mt][v] + rb);
        xi_o[off] = acc[1][j][mt][v] * sigmoidf_(acc[2][j][mt][v] + igb);
        og_o[off] = sigmoidf_(acc[3][j][mt][v] + ogb);
      }
    }
  }
}

// ===========================================================================
// 3-pass chunked linear recurrence  h_t = h_{t-1}*r_t + xi_t
// ===========================================================================
__global__ void sqrll_scan_pass1(const float* __restrict__ xi, const float* __restrict__ r,
                                 float* __restrict__ Ac, float* __restrict__ Bc)
{
  const int gid = blockIdx.x * blockDim.x + threadIdx.x;   // B*NCHUNK*NM threads
  const int m = gid & (NM_ - 1);
  const int c = (gid >> 10) & (NCHUNK - 1);
  const int b = gid >> 14;
  size_t off = (size_t)(b * T_ + c * CHUNK) * NM_ + m;
  float h = 0.0f, A = 1.0f;
  for (int t = 0; t < CHUNK; ++t) {
    const float rv = r[off];
    h = fmaf(h, rv, xi[off]);
    A *= rv;
    off += NM_;
  }
  const size_t coff = (size_t)(b * NCHUNK + c) * NM_ + m;
  Ac[coff] = A;
  Bc[coff] = h;
}

__global__ void sqrll_scan_pass2(const float* __restrict__ Ac, const float* __restrict__ Bc,
                                 const float* __restrict__ mem0,
                                 float* __restrict__ Cc, float* __restrict__ mem_out)
{
  const int gid = blockIdx.x * blockDim.x + threadIdx.x;   // B*NM threads
  const int m = gid & (NM_ - 1);
  const int b = gid >> 10;
  float carry = mem0[(size_t)b * NM_ + m];
  #pragma unroll
  for (int c = 0; c < NCHUNK; ++c) {
    const size_t coff = (size_t)(b * NCHUNK + c) * NM_ + m;
    Cc[coff] = carry;                         // h entering chunk c
    carry = fmaf(carry, Ac[coff], Bc[coff]);
  }
  mem_out[(size_t)b * NM_ + m] = carry;       // == y[:, -1]
}

__global__ void sqrll_scan_pass3(float* __restrict__ xi_y, const float* __restrict__ r,
                                 const float* __restrict__ Cc)
{
  const int gid = blockIdx.x * blockDim.x + threadIdx.x;
  const int m = gid & (NM_ - 1);
  const int c = (gid >> 10) & (NCHUNK - 1);
  const int b = gid >> 14;
  size_t off = (size_t)(b * T_ + c * CHUNK) * NM_ + m;
  float h = Cc[(size_t)(b * NCHUNK + c) * NM_ + m];
  for (int t = 0; t < CHUNK; ++t) {
    const float rv = r[off];
    h = fmaf(h, rv, xi_y[off]);
    xi_y[off] = h;                            // y written in place over xi
    off += NM_;
  }
}

// ===========================================================================
// Kernel 3: output GEMM with fused softsign(y)*og on the A side.
// (Staging needs elementwise math, so it keeps the VALU-cvt path.)
// ===========================================================================
#define GO_BM 128
#define GO_BN 64
#define GO_BK 32
#define GO_PITCH 40

__launch_bounds__(256, 2)
__global__ void sqrll_out(const float* __restrict__ y, const float* __restrict__ og,
                          const float* __restrict__ wo_w, float* __restrict__ out)
{
  __shared__ __align__(16) bf16_t As[GO_BM * GO_PITCH];
  __shared__ __align__(16) bf16_t Bs[GO_BN * GO_PITCH];

  const int m0   = blockIdx.x * GO_BM;
  const int n0   = blockIdx.y * GO_BN;
  const int tid  = threadIdx.x;
  const int wv   = tid >> 5;
  const int lane = tid & 31;
  const int hlf  = lane >> 4;
  const int l16  = lane & 15;

  v8f acc[4] = {};                     // 4 N-tiles of 16

  for (int k0 = 0; k0 < NM_; k0 += GO_BK) {
    __syncthreads();
    #pragma unroll
    for (int i = 0; i < 4; ++i) {
      const int chunk = tid + i * 256;
      const int row = chunk >> 3, kq = chunk & 7;
      const size_t goff = (size_t)(m0 + row) * NM_ + k0 + kq * 4;
      const float4 yv = *(const float4*)(y + goff);
      const float4 gv = *(const float4*)(og + goff);
      const float s0 = yv.x / (1.0f + fabsf(yv.x)) * gv.x;
      const float s1 = yv.y / (1.0f + fabsf(yv.y)) * gv.y;
      const float s2 = yv.z / (1.0f + fabsf(yv.z)) * gv.z;
      const float s3 = yv.w / (1.0f + fabsf(yv.w)) * gv.w;
      bf16x4 bv = { (bf16_t)s0, (bf16_t)s1, (bf16_t)s2, (bf16_t)s3 };
      *(bf16x4*)(&As[row * GO_PITCH + kq * 4]) = bv;
    }
    #pragma unroll
    for (int i = 0; i < 2; ++i) {
      const int chunk = tid + i * 256;          // 64 rows x 8 = 512
      const int row = chunk >> 3, kq = chunk & 7;
      const float4 v = *(const float4*)(wo_w + (size_t)(n0 + row) * NM_ + k0 + kq * 4);
      bf16x4 bv = { (bf16_t)v.x, (bf16_t)v.y, (bf16_t)v.z, (bf16_t)v.w };
      *(bf16x4*)(&Bs[row * GO_PITCH + kq * 4]) = bv;
    }
    __syncthreads();

    const v16bf a = load_frag_a(&As[(wv * 16 + l16) * GO_PITCH], hlf);
    #pragma unroll
    for (int j = 0; j < 4; ++j) {
      const v16bf b = load_frag_b(&Bs[(j * 16 + l16) * GO_PITCH], hlf);
      acc[j] = __builtin_amdgcn_wmma_f32_16x16x32_bf16(
          false, a, false, b, (short)0, acc[j], false, false);
    }
  }

  #pragma unroll
  for (int j = 0; j < 4; ++j) {
    const int n = n0 + j * 16 + l16;
    #pragma unroll
    for (int v = 0; v < 8; ++v) {
      const int m = m0 + wv * 16 + hlf * 8 + v;
      out[(size_t)m * NOUT_ + n] = acc[j][v];
    }
  }
}

// ===========================================================================
extern "C" void kernel_launch(void* const* d_in, const int* in_sizes, int n_in,
                              void* d_out, int out_size, void* d_ws, size_t ws_size,
                              hipStream_t stream)
{
  (void)in_sizes; (void)n_in; (void)out_size; (void)ws_size;
  const float* x     = (const float*)d_in[0];
  const float* mem0  = (const float*)d_in[1];
  const float* wr_w  = (const float*)d_in[2];
  const float* wr_b  = (const float*)d_in[3];
  const float* wi_w  = (const float*)d_in[4];
  const float* wig_w = (const float*)d_in[5];
  const float* wig_b = (const float*)d_in[6];
  const float* wog_w = (const float*)d_in[7];
  const float* wog_b = (const float*)d_in[8];
  const float* wo_w  = (const float*)d_in[9];

  float* out     = (float*)d_out;                       // y [B,T,NOUT] flat
  float* mem_out = out + (size_t)MT_ * NOUT_;           // then mem_out [B,NM]

  // workspace: xi | r | og (f32) | Ac | Bc | Cc | Xb (bf16) | Wb (bf16 x4)
  float* xi = (float*)d_ws;
  float* r  = xi + (size_t)MT_ * NM_;
  float* og = r  + (size_t)MT_ * NM_;
  float* Ac = og + (size_t)MT_ * NM_;
  float* Bc = Ac + (size_t)B_ * NCHUNK * NM_;
  float* Cc = Bc + (size_t)B_ * NCHUNK * NM_;
  bf16_t* Xb = (bf16_t*)(Cc + (size_t)B_ * NCHUNK * NM_);
  bf16_t* Wb = Xb + (size_t)MT_ * NIN_;                 // [4][NM_][NIN_]

  // one-shot f32 -> bf16 conversions (x + 4 gate weight matrices)
  cvt_f32_to_bf16<<<(MT_ * NIN_ / 4) / 256, 256, 0, stream>>>(x, Xb, MT_ * NIN_ / 4);
  const float* gw[4] = { wr_w, wi_w, wig_w, wog_w };
  for (int g = 0; g < 4; ++g)
    cvt_f32_to_bf16<<<(NM_ * NIN_ / 4) / 256, 256, 0, stream>>>(
        gw[g], Wb + (size_t)g * NM_ * NIN_, NM_ * NIN_ / 4);

  sqrll_gates<<<dim3(MT_ / GA_BM, NM_ / GA_BN), 256, 0, stream>>>(
      Xb, Wb, wr_b, wig_b, wog_b, xi, r, og);

  sqrll_scan_pass1<<<(B_ * NM_ * NCHUNK) / 256, 256, 0, stream>>>(xi, r, Ac, Bc);
  sqrll_scan_pass2<<<(B_ * NM_) / 256, 256, 0, stream>>>(Ac, Bc, mem0, Cc, mem_out);
  sqrll_scan_pass3<<<(B_ * NM_ * NCHUNK) / 256, 256, 0, stream>>>(xi, r, Cc);

  sqrll_out<<<dim3(MT_ / GO_BM, NOUT_ / GO_BN), 256, 0, stream>>>(xi, og, wo_w, out);
}
